// HLG_Old_80650895884471
// MI455X (gfx1250) — matrix operand order repaired and checked by reading, not compiled, not run on hardware
//
#include <hip/hip_runtime.h>
#include <hip/hip_bf16.h>

typedef __attribute__((ext_vector_type(16))) _Float16 v16h;
typedef __attribute__((ext_vector_type(8)))  float    v8f;

#define HH 128

// ---------------------------------------------------------------------------
// Pack W[K,128] (f32, row-major) into WMMA B-fragment order (f16):
//   Wp[(((kb*8 + nt)*32 + lane)*16) + e] = W[(kb*32 + 16*(lane>>4) + e)*128 + nt*16 + (lane&15)]
// so each lane's 16 B elements for a (kb, nt) tile are 32 contiguous bytes.
// ---------------------------------------------------------------------------
__global__ void pack_w_f16(const float* __restrict__ W, _Float16* __restrict__ Wp, int nk)
{
    const int t = blockIdx.x * blockDim.x + threadIdx.x;
    const int total = nk * 8 * 32;
    if (t >= total) return;
    const int lane = t & 31;
    const int nt   = (t >> 5) & 7;
    const int kb   = t >> 8;
    const int kh   = lane >> 4;
    const int nn   = lane & 15;
    const float* src = W + (size_t)(kb * 32 + 16 * kh) * HH + nt * 16 + nn;
    _Float16* dst = Wp + (size_t)t * 16;
#pragma unroll
    for (int e = 0; e < 16; ++e) dst[e] = (_Float16)src[(size_t)e * HH];
}

// ---------------------------------------------------------------------------
// WMMA GEMM: out[rows,128] = act( concat(A0[,A1[,A2]])[rows, segs*128] @ W + b )
// W comes pre-packed (f16 fragment order). Each 128-K segment (32KB) is staged
// in LDS by the whole block, then all 4 waves consume it via ds_load while
// streaming A from global (b128 loads + f16 packing) into
// v_wmma_f32_16x16x32_f16. Segment-0 rows may be gathered through g0.
// ---------------------------------------------------------------------------
__global__ void wmma_gemm128(const float* __restrict__ A0, const float* __restrict__ A1,
                             const float* __restrict__ A2, const int* __restrict__ g0,
                             const _Float16* __restrict__ Wp, const float* __restrict__ bias,
                             float* __restrict__ out, int rows, int segs, int relu)
{
    __shared__ alignas(32) _Float16 smem[16384];   // one 128-K segment of packed W (32 KB)
    const int tid  = threadIdx.x;
    const int lane = tid & 31;
    const int wid  = tid >> 5;
    const int row0 = (blockIdx.x * 4 + wid) * 16;
    const bool active = (row0 < rows);            // wave-uniform; EXEC all-1 inside
    const int m  = lane & 15;                     // A-row within tile
    const int kh = lane >> 4;                     // k-half: lane holds K = 16*kh + [0..15] (this code's order)
    const int nn = lane & 15;                     // D column within 16-wide tile
    const int r  = active ? (row0 + m) : 0;
    const int rg = (active && g0) ? g0[r] : r;

    v8f acc[8];
#pragma unroll
    for (int i = 0; i < 8; ++i)
#pragma unroll
        for (int j = 0; j < 8; ++j) acc[i][j] = 0.0f;

    const float* As[3] = {A0, A1, A2};
    for (int seg = 0; seg < segs; ++seg) {
        // ---- stage this segment's packed W into LDS (block-cooperative) ----
        __syncthreads();                                   // protect previous reads
        {
            const float4* src4 = (const float4*)(Wp + (size_t)seg * 16384);
            float4* dst4 = (float4*)smem;
#pragma unroll
            for (int i = 0; i < 16; ++i) dst4[tid + i * 128] = src4[tid + i * 128];
        }
        __syncthreads();

        if (active) {
            const int ar = (seg == 0) ? rg : r;
            const float* Abase = As[seg] + (size_t)ar * HH;
#pragma unroll
            for (int kb = 0; kb < 4; ++kb) {
                const float* Ap = Abase + kb * 32;
                if (kb < 3) __builtin_prefetch(Ap + 32, 0, 1);     // global_prefetch
                // A fragment: lanes 0-15 hold K=[0..15]; lanes 16-31 hold K=[16..31]
                const float4* p0 = (const float4*)(Ap + 16 * kh);
                const float4* p1 = (const float4*)(Ap + 16 * kh + 8);
                const float4 q0 = p0[0], q1 = p0[1], q2 = p1[0], q3 = p1[1];
                v16h a;
                a[0]  = (_Float16)q0.x; a[1]  = (_Float16)q0.y; a[2]  = (_Float16)q0.z; a[3]  = (_Float16)q0.w;
                a[4]  = (_Float16)q1.x; a[5]  = (_Float16)q1.y; a[6]  = (_Float16)q1.z; a[7]  = (_Float16)q1.w;
                a[8]  = (_Float16)q2.x; a[9]  = (_Float16)q2.y; a[10] = (_Float16)q2.z; a[11] = (_Float16)q2.w;
                a[12] = (_Float16)q3.x; a[13] = (_Float16)q3.y; a[14] = (_Float16)q3.z; a[15] = (_Float16)q3.w;
#pragma unroll
                for (int nt = 0; nt < 8; ++nt) {
                    const v16h b = *(const v16h*)(smem + (size_t)(((kb * 8 + nt) * 32) + lane) * 16);
                    acc[nt] = __builtin_amdgcn_wmma_f32_16x16x32_f16(
                        false, a, false, b, (short)0, acc[nt], false, false);
                }
            }
        }
    }

    if (active) {
        // Epilogue: D element e -> M = e + 8*kh, col = nt*16 + nn
#pragma unroll
        for (int nt = 0; nt < 8; ++nt) {
            const int col = nt * 16 + nn;
            const float bv = bias ? bias[col] : 0.0f;
#pragma unroll
            for (int e = 0; e < 8; ++e) {
                float v = acc[nt][e] + bv;
                if (relu) v = fmaxf(v, 0.0f);
                out[(size_t)(row0 + e + 8 * kh) * HH + col] = v;
            }
        }
    }
}

// ---------------------------------------------------------------------------
// Elementwise / scatter / BN helper kernels (fp32, memory-bound path)
// ---------------------------------------------------------------------------
__global__ void zero_f32(float* __restrict__ p, long long n)
{
    long long i = (long long)blockIdx.x * blockDim.x + threadIdx.x;
    if (i < n) p[i] = 0.0f;
}

// out[i,:] = sum_f emb[f, feat[i,f], :]
__global__ void cat_encode(const float* __restrict__ emb, const int* __restrict__ feat,
                           float* __restrict__ out, int nf, int vocab)
{
    const int i = blockIdx.x, h = threadIdx.x;
    float s = 0.0f;
    for (int f = 0; f < nf; ++f) {
        const int t = feat[(size_t)i * nf + f];
        s += emb[((size_t)f * vocab + t) * HH + h];
    }
    out[(size_t)i * HH + h] = s;
}

// sums[dmap?dmap[didx[i]]:didx[i], :] += src[sidx?sidx[i]:i, :]; cnt += 1
__global__ void scatter_add128(const float* __restrict__ src, const int* __restrict__ sidx,
                               const int* __restrict__ didx, const int* __restrict__ dmap,
                               float* __restrict__ sums, float* __restrict__ cnt, int Msrc)
{
    const int i = blockIdx.x, h = threadIdx.x;
    if (i >= Msrc) return;
    const int s = sidx ? sidx[i] : i;
    int d = didx[i];
    if (dmap) d = dmap[d];
    atomicAdd(&sums[(size_t)d * HH + h], src[(size_t)s * HH + h]);
    if (h == 0) atomicAdd(&cnt[d], 1.0f);
}

__global__ void seg_div128(float* __restrict__ sums, const float* __restrict__ cnt)
{
    const int i = blockIdx.x, h = threadIdx.x;
    const float c = fmaxf(cnt[i], 1.0f);
    sums[(size_t)i * HH + h] /= c;
}

// out[i,:] = 0.5*(x[r[i],:] + x[c[i],:])
__global__ void edge_avg(const float* __restrict__ x, const int* __restrict__ r,
                         const int* __restrict__ c, float* __restrict__ out)
{
    const int i = blockIdx.x, h = threadIdx.x;
    out[(size_t)i * HH + h] = 0.5f * (x[(size_t)r[i] * HH + h] + x[(size_t)c[i] * HH + h]);
}

// Per-column partial sum/sumsq over a 256-row slab (thread == column: coalesced)
__global__ void bn_stats(const float* __restrict__ A, int rows, float* __restrict__ sum,
                         float* __restrict__ sumsq)
{
    const int c  = threadIdx.x;
    const int r0 = blockIdx.x * 256;
    int r1 = r0 + 256; if (r1 > rows) r1 = rows;
    float s = 0.0f, q = 0.0f;
    for (int r = r0; r < r1; ++r) {
        const float v = A[(size_t)r * HH + c];
        s += v; q += v * v;
    }
    atomicAdd(&sum[c], s);
    atomicAdd(&sumsq[c], q);
}

__global__ void bn_finalize(const float* __restrict__ sum, const float* __restrict__ sumsq,
                            int rows, float* __restrict__ mean, float* __restrict__ rstd)
{
    const int c = threadIdx.x;
    const float inv = 1.0f / (float)rows;
    const float mu  = sum[c] * inv;
    float var = sumsq[c] * inv - mu * mu;
    if (var < 0.0f) var = 0.0f;
    mean[c] = mu;
    rstd[c] = rsqrtf(var + 1e-5f);
}

// out = (resid?resid:0) + relu((t - mean)*rstd*g + beta)
__global__ void bn_apply(const float* __restrict__ t, const float* __restrict__ mean,
                         const float* __restrict__ rstd, const float* __restrict__ g,
                         const float* __restrict__ beta, const float* __restrict__ resid,
                         float* __restrict__ out, long long total)
{
    long long i = (long long)blockIdx.x * blockDim.x + threadIdx.x;
    if (i >= total) return;
    const int c = (int)(i & (HH - 1));
    float v = (t[i] - mean[c]) * rstd[c] * g[c] + beta[c];
    v = fmaxf(v, 0.0f);
    if (resid) v += resid[i];
    out[i] = v;
}

__global__ void vec_add(float* __restrict__ a, const float* __restrict__ b, long long n)
{
    long long i = (long long)blockIdx.x * blockDim.x + threadIdx.x;
    if (i < n) a[i] += b[i];
}

// out[i] = xg[i,:] . W[:,0] + b[0]
__global__ void final_out(const float* __restrict__ xg, const float* __restrict__ W,
                          const float* __restrict__ b, float* __restrict__ out, int rows)
{
    const int i = blockIdx.x * blockDim.x + threadIdx.x;
    if (i >= rows) return;
    float s = b[0];
    for (int k = 0; k < HH; ++k) s += xg[(size_t)i * HH + k] * W[k];
    out[i] = s;
}

// ---------------------------------------------------------------------------
extern "C" void kernel_launch(void* const* d_in, const int* in_sizes, int n_in,
                              void* d_out, int out_size, void* d_ws, size_t ws_size,
                              hipStream_t stream)
{
    (void)n_in; (void)ws_size;
    const float* P        = (const float*)d_in[0];
    const int* x_atom     = (const int*)d_in[1];
    const int* edge_attr  = (const int*)d_in[2];
    const int* frag_types = (const int*)d_in[3];
    const int* eidx       = (const int*)d_in[4];
    const int* fidx       = (const int*)d_in[5];
    const int* lhidx      = (const int*)d_in[6];
    const int* hidx       = (const int*)d_in[7];
    const int* batch      = (const int*)d_in[8];
    const int* fbatch     = (const int*)d_in[9];

    const int Nn  = in_sizes[8];
    const int Ee  = in_sizes[2] / 3;
    const int Ff  = in_sizes[3];
    const int Mm  = in_sizes[5] / 2;
    const int M2m = in_sizes[6] / 2;
    const int M3m = in_sizes[7] / 2;
    const int Bb  = out_size;

    const int* erow = eidx;       const int* ecol = eidx + Ee;
    const int* frow = fidx;       const int* fcol = fidx + Mm;
    const int* lhe  = lhidx;      const int* lhf  = lhidx + M2m;
    const int* hrow = hidx;       const int* hcol = hidx + M3m;

    // ---- params: JAX pytree flattening => dict keys in sorted order ----
    size_t cur = 0;
    auto take = [&](size_t n) { const float* p = P + cur; cur += n; return p; };
    struct OutMLP { const float *W, *b, *beta, *g; };
    auto takeOut = [&]() {
        OutMLP o; o.W = take(HH * HH); o.b = take(HH); o.beta = take(HH); o.g = take(HH); return o;
    };
    OutMLP ao0 = takeOut(), ao1 = takeOut();                 // ao0_*, ao1_*
    const float* atom_emb = take((size_t)9 * 128 * HH);      // atom_emb
    const float* bond_emb = take((size_t)3 * 16 * HH);       // bond_emb
    OutMLP eo0 = takeOut(), eo1 = takeOut();                 // eo0_*, eo1_*
    OutMLP fo0 = takeOut(), fo1 = takeOut();                 // fo0_*, fo1_*
    const float* frag_emb = take((size_t)120 * HH);          // frag_emb
    struct Lay {
        const float *a2a0W,*a2a0b,*a2a1W,*a2a1b,*abW,*abb;
        const float *a2e0W,*a2e0b,*a2e1W,*a2e1b;
        const float *a2f0W,*a2f0b,*a2f1W,*a2f1b;
        const float *bnab,*bnag,*bneb,*bneg,*bnfb,*bnfg;
        const float *caW,*cab,*cabeta,*cag;
        const float *ceW,*ceb,*cebeta,*ceg;
        const float *cfW,*cfb,*cfbeta,*cfg;
        const float *e2f0W,*e2f0b,*e2f1W,*e2f1b;
        const float *f2a0W,*f2a0b,*f2a1W,*f2a1b;
        const float *f2f0W,*f2f0b,*f2f1W,*f2f1b;
    } Lyr[2];
    for (int l = 0; l < 2; ++l) {                            // l{l}_* keys, sorted
        Lay& L = Lyr[l];
        L.a2a0W=take(HH*HH); L.a2a0b=take(HH); L.a2a1W=take(HH*HH); L.a2a1b=take(HH);
        L.abW=take(2*HH*HH); L.abb=take(HH);
        L.a2e0W=take(HH*HH); L.a2e0b=take(HH); L.a2e1W=take(HH*HH); L.a2e1b=take(HH);
        L.a2f0W=take(HH*HH); L.a2f0b=take(HH); L.a2f1W=take(HH*HH); L.a2f1b=take(HH);
        L.bnab=take(HH); L.bnag=take(HH); L.bneb=take(HH); L.bneg=take(HH);
        L.bnfb=take(HH); L.bnfg=take(HH);
        L.caW=take(2*HH*HH); L.cab=take(HH); L.cabeta=take(HH); L.cag=take(HH);
        L.ceW=take(HH*HH);   L.ceb=take(HH); L.cebeta=take(HH); L.ceg=take(HH);
        L.cfW=take(3*HH*HH); L.cfb=take(HH); L.cfbeta=take(HH); L.cfg=take(HH);
        L.e2f0W=take(HH*HH); L.e2f0b=take(HH); L.e2f1W=take(HH*HH); L.e2f1b=take(HH);
        L.f2a0W=take(HH*HH); L.f2a0b=take(HH); L.f2a1W=take(HH*HH); L.f2a1b=take(HH);
        L.f2f0W=take(HH*HH); L.f2f0b=take(HH); L.f2f1W=take(HH*HH); L.f2f1b=take(HH);
    }
    const float* outW = take(HH);
    const float* outb = take(1);

    // ---- workspace partition (256B aligned) ----
    float* w = (float*)d_ws;
    size_t off = 0;
    auto wsalloc = [&](size_t nfl) { float* p = w + off; off += (nfl + 63) & ~(size_t)63; return p; };
    float* x   = wsalloc((size_t)Nn * HH);
    float* xe  = wsalloc((size_t)Ee * HH);
    float* xf  = wsalloc((size_t)Ff * HH);
    float* e0  = wsalloc((size_t)Ee * HH);
    float* e1  = wsalloc((size_t)Ee * HH);
    float* na  = wsalloc((size_t)Nn * HH);
    float* nb  = wsalloc((size_t)Nn * HH);
    float* nc  = wsalloc((size_t)Nn * HH);
    float* fa  = wsalloc((size_t)Ff * HH);
    float* fb  = wsalloc((size_t)Ff * HH);
    float* fc  = wsalloc((size_t)Ff * HH);
    float* fd  = wsalloc((size_t)Ff * HH);
    float* ga  = wsalloc((size_t)Bb * HH);
    float* gb  = wsalloc((size_t)Bb * HH);
    float* gacc= wsalloc((size_t)Bb * HH);
    float* cnt = wsalloc((size_t)(Nn > Ff ? Nn : Ff));
    float* bnst= wsalloc(256);                      // [sum | sumsq]
    float* bnmr= wsalloc(256);                      // [mean | rstd]
    _Float16* wpack = (_Float16*)wsalloc(24576);    // 3*128*128 f16 packed weights

    // ---- stream-side helpers ----
    auto zero = [&](float* p, size_t nfl) {
        hipLaunchKernelGGL(zero_f32, dim3((unsigned)((nfl + 255) / 256)), dim3(256), 0, stream,
                           p, (long long)nfl);
    };
    auto gemm = [&](const float* A0, const float* A1, const float* A2, const int* g0,
                    const float* Wt, const float* bias, float* outp, int rows, int segs, int relu) {
        const int nk = segs * 4;
        hipLaunchKernelGGL(pack_w_f16, dim3((nk * 256 + 127) / 128), dim3(128), 0, stream,
                           Wt, wpack, nk);
        const int blocks = ((rows + 15) / 16 + 3) / 4;
        hipLaunchKernelGGL(wmma_gemm128, dim3(blocks), dim3(128), 0, stream,
                           A0, A1, A2, g0, wpack, bias, outp, rows, segs, relu);
    };
    auto smean = [&](const float* src, const int* sidx, const int* didx, const int* dmap,
                     int Msrc, float* dst, int nseg) {
        zero(dst, (size_t)nseg * HH);
        zero(cnt, (size_t)nseg);
        hipLaunchKernelGGL(scatter_add128, dim3(Msrc), dim3(HH), 0, stream,
                           src, sidx, didx, dmap, dst, cnt, Msrc);
        hipLaunchKernelGGL(seg_div128, dim3(nseg), dim3(HH), 0, stream, dst, cnt);
    };
    auto bnrelu = [&](const float* t, int rows, const float* g, const float* beta,
                      const float* resid, float* outp) {
        zero(bnst, 256);
        hipLaunchKernelGGL(bn_stats, dim3((rows + 255) / 256), dim3(HH), 0, stream,
                           t, rows, bnst, bnst + HH);
        hipLaunchKernelGGL(bn_finalize, dim3(1), dim3(HH), 0, stream,
                           bnst, bnst + HH, rows, bnmr, bnmr + HH);
        const long long tot = (long long)rows * HH;
        hipLaunchKernelGGL(bn_apply, dim3((unsigned)((tot + 255) / 256)), dim3(256), 0, stream,
                           t, bnmr, bnmr + HH, g, beta, resid, outp, tot);
    };

    // ---- encoders ----
    hipLaunchKernelGGL(cat_encode, dim3(Nn), dim3(HH), 0, stream, atom_emb, x_atom, x, 9, 128);
    hipLaunchKernelGGL(cat_encode, dim3(Ee), dim3(HH), 0, stream, bond_emb, edge_attr, xe, 3, 16);
    hipLaunchKernelGGL(cat_encode, dim3(Ff), dim3(HH), 0, stream, frag_emb, frag_types, xf, 1, 120);

    // ---- message-passing layers ----
    for (int l = 0; l < 2; ++l) {
        const Lay& L = Lyr[l];
        // atom2atom: relu(concat(x[re], xe) @ Wb) -> smean over targets -> 2x(Lin+ReLU)
        gemm(x, xe, nullptr, erow, L.abW, L.abb, e0, Ee, 2, 1);
        smean(e0, nullptr, ecol, nullptr, Ee, na, Nn);
        gemm(na, nullptr, nullptr, nullptr, L.a2a0W, L.a2a0b, nb, Nn, 1, 1);
        gemm(nb, nullptr, nullptr, nullptr, L.a2a1W, L.a2a1b, na, Nn, 1, 1);   // m_a2a -> na
        // frag2atom
        smean(xf, fcol, frow, nullptr, Mm, nb, Nn);
        gemm(nb, nullptr, nullptr, nullptr, L.f2a0W, L.f2a0b, nc, Nn, 1, 1);
        gemm(nc, nullptr, nullptr, nullptr, L.f2a1W, L.f2a1b, nb, Nn, 1, 1);   // m_f2a -> nb
        // combine atom + BN stacks
        gemm(na, nb, nullptr, nullptr, L.caW, L.cab, nc, Nn, 2, 0);
        bnrelu(nc, Nn, L.cag, L.cabeta, x, x);       // x = x + relu(bn(t))
        bnrelu(x, Nn, L.bnag, L.bnab, nullptr, x);   // x = relu(bn(x))
        // atom2edge
        hipLaunchKernelGGL(edge_avg, dim3(Ee), dim3(HH), 0, stream, x, erow, ecol, e0);
        gemm(e0, nullptr, nullptr, nullptr, L.a2e0W, L.a2e0b, e1, Ee, 1, 1);
        gemm(e1, nullptr, nullptr, nullptr, L.a2e1W, L.a2e1b, e0, Ee, 1, 1);
        gemm(e0, nullptr, nullptr, nullptr, L.ceW, L.ceb, e1, Ee, 1, 0);
        bnrelu(e1, Ee, L.ceg, L.cebeta, xe, xe);
        bnrelu(xe, Ee, L.bneg, L.bneb, nullptr, xe);
        // atom2frag
        smean(x, frow, fcol, nullptr, Mm, fa, Ff);
        gemm(fa, nullptr, nullptr, nullptr, L.a2f0W, L.a2f0b, fd, Ff, 1, 1);
        gemm(fd, nullptr, nullptr, nullptr, L.a2f1W, L.a2f1b, fa, Ff, 1, 1);
        // edge2frag
        smean(xe, lhe, lhf, nullptr, M2m, fb, Ff);
        gemm(fb, nullptr, nullptr, nullptr, L.e2f0W, L.e2f0b, fd, Ff, 1, 1);
        gemm(fd, nullptr, nullptr, nullptr, L.e2f1W, L.e2f1b, fb, Ff, 1, 1);
        // frag2frag
        smean(xf, hrow, hcol, nullptr, M3m, fc, Ff);
        gemm(fc, nullptr, nullptr, nullptr, L.f2f0W, L.f2f0b, fd, Ff, 1, 1);
        gemm(fd, nullptr, nullptr, nullptr, L.f2f1W, L.f2f1b, fc, Ff, 1, 1);
        // combine frag + BN stacks
        gemm(fa, fb, fc, nullptr, L.cfW, L.cfb, fd, Ff, 3, 0);
        bnrelu(fd, Ff, L.cfg, L.cfbeta, xf, xf);
        bnrelu(xf, Ff, L.bnfg, L.bnfb, nullptr, xf);
    }

    // ---- readout ----
    const long long gtot = (long long)Bb * HH;
    smean(x, nullptr, batch, nullptr, Nn, ga, Bb);
    gemm(ga, nullptr, nullptr, nullptr, ao0.W, ao0.b, gb, Bb, 1, 0);
    bnrelu(gb, Bb, ao0.g, ao0.beta, nullptr, ga);
    gemm(ga, nullptr, nullptr, nullptr, ao1.W, ao1.b, gb, Bb, 1, 0);
    bnrelu(gb, Bb, ao1.g, ao1.beta, nullptr, gacc);                    // xg

    smean(xe, nullptr, erow, batch, Ee, ga, Bb);                       // idx = batch[re]
    gemm(ga, nullptr, nullptr, nullptr, eo0.W, eo0.b, gb, Bb, 1, 0);
    bnrelu(gb, Bb, eo0.g, eo0.beta, nullptr, ga);
    gemm(ga, nullptr, nullptr, nullptr, eo1.W, eo1.b, gb, Bb, 1, 0);
    bnrelu(gb, Bb, eo1.g, eo1.beta, nullptr, ga);
    hipLaunchKernelGGL(vec_add, dim3((unsigned)((gtot + 255) / 256)), dim3(256), 0, stream,
                       gacc, ga, gtot);

    smean(xf, nullptr, fbatch, nullptr, Ff, ga, Bb);
    gemm(ga, nullptr, nullptr, nullptr, fo0.W, fo0.b, gb, Bb, 1, 0);
    bnrelu(gb, Bb, fo0.g, fo0.beta, nullptr, ga);
    gemm(ga, nullptr, nullptr, nullptr, fo1.W, fo1.b, gb, Bb, 1, 0);
    bnrelu(gb, Bb, fo1.g, fo1.beta, nullptr, ga);
    hipLaunchKernelGGL(vec_add, dim3((unsigned)((gtot + 255) / 256)), dim3(256), 0, stream,
                       gacc, ga, gtot);

    hipLaunchKernelGGL(final_out, dim3((Bb + 255) / 256), dim3(256), 0, stream,
                       gacc, outW, outb, (float*)d_out, Bb);
}
